// GAT_44126493999472
// MI455X (gfx1250) — compile-verified, matrix operand order checked by reading, not run on hardware
//
#include <hip/hip_runtime.h>
#include <hip/hip_bf16.h>

// ---------------------------------------------------------------------------
// GAT (2-layer) for MI455X / gfx1250.
// Dense node transforms: v_wmma_f32_16x16x32_bf16, f32 accumulate.
//   - one wave computes a 16x64 output slab: 4 accumulators reuse each
//     A-fragment across 4 WMMAs (A traffic & bf16 converts amortized 4x).
// Graph softmax/aggregation: u32-key atomicMax + f32 atomicAdd; edge logits
// materialized once (streaming) instead of re-gathered per pass.
// ---------------------------------------------------------------------------

typedef __attribute__((ext_vector_type(16))) __bf16 v16bf;
typedef __attribute__((ext_vector_type(8)))  float  v8f;

constexpr int NNODE = 50000;   // multiple of 16 (3125 row tiles)
constexpr int NEDGE = 800000;
constexpr int FIN   = 512;
constexpr int H1    = 8;
constexpr int FHID  = 8;
constexpr int HID   = H1 * FHID;   // 64
constexpr int NLAB  = 64;
constexpr int ROWT  = NNODE / 16;  // 3125
constexpr float SLOPE = 0.2f;

#define DEVINL __device__ __forceinline__

// Monotone order-preserving float -> u32 key (for atomicMax-based segment max).
DEVINL unsigned fkey(float v) {
    unsigned u = __float_as_uint(v);
    return (u & 0x80000000u) ? ~u : (u | 0x80000000u);
}
DEVINL float funkey(unsigned k) {
    return (k & 0x80000000u) ? __uint_as_float(k ^ 0x80000000u)
                             : __uint_as_float(~k);
}
DEVINL float lrelu(float v) { return v > 0.f ? v : SLOPE * v; }

// ---------------------------------------------------------------------------
// Repack weights into the CDNA5 bf16 B-operand lane layout.
// Bp index = ((nt*nKt + kt)*32 + lane)*16 + e, element e of lane L holds
// B[kt*32 + K(e,L)][nt*16 + L%16],  K(e,L) = (e<8?0:16) + (L/16)*8 + (e%8).
// gat1: B[k][n] = W1[n/8][k][n%8]   else: B[k][n] = W2[0][k][n]
// ---------------------------------------------------------------------------
__global__ void k_repack(const float* __restrict__ W, __bf16* __restrict__ Bp,
                         int nKt, int total, int gat1) {
    int idx = blockIdx.x * blockDim.x + threadIdx.x;
    if (idx >= total) return;
    int e    = idx & 15;
    int lane = (idx >> 4) & 31;
    int ktnt = idx >> 9;
    int kt   = ktnt % nKt;
    int nt   = ktnt / nKt;
    int K = ((e < 8) ? 0 : 16) + (lane >> 4) * 8 + (e & 7);
    int k = kt * 32 + K;
    int n = nt * 16 + (lane & 15);
    float v = gat1 ? W[(size_t)(n >> 3) * (FIN * FHID) + (size_t)k * FHID + (n & 7)]
                   : W[(size_t)k * NLAB + n];
    Bp[idx] = (__bf16)v;
}

// ---------------------------------------------------------------------------
// WMMA GEMM: C[NNODE x 64] = A[NNODE x lda (fp32)] * Bpack (bf16)
// One wave per 16-row tile; 4 f32 accumulators cover all 64 columns, so each
// A fragment feeds 4 back-to-back WMMAs.
// ---------------------------------------------------------------------------
__global__ void __launch_bounds__(128)
k_gemm_bf16(const float* __restrict__ A, int lda,
            const __bf16* __restrict__ Bp, int nKt,
            float* __restrict__ C) {
    int lane = threadIdx.x & 31;
    int wave = threadIdx.x >> 5;
    int tile = blockIdx.x * 4 + wave;
    if (tile >= ROWT) return;          // whole-wave exit; active waves keep EXEC all-1s

    int m     = lane & 15;
    int khalf = lane >> 4;
    const float*  arow = A  + (size_t)(tile * 16 + m) * lda + khalf * 8;
    const __bf16* bl   = Bp + (size_t)lane * 16;

    v8f acc[4] = {};
    for (int kt = 0; kt < nKt; ++kt) {
        const float* ap = arow + kt * 32;
        float4 a0 = *(const float4*)(ap);
        float4 a1 = *(const float4*)(ap + 4);
        float4 a2 = *(const float4*)(ap + 16);
        float4 a3 = *(const float4*)(ap + 20);
        if (kt + 1 < nKt) __builtin_prefetch(ap + 32, 0, 0);   // global_prefetch_b8

        v16bf av;
        av[0]  = (__bf16)a0.x; av[1]  = (__bf16)a0.y; av[2]  = (__bf16)a0.z; av[3]  = (__bf16)a0.w;
        av[4]  = (__bf16)a1.x; av[5]  = (__bf16)a1.y; av[6]  = (__bf16)a1.z; av[7]  = (__bf16)a1.w;
        av[8]  = (__bf16)a2.x; av[9]  = (__bf16)a2.y; av[10] = (__bf16)a2.z; av[11] = (__bf16)a2.w;
        av[12] = (__bf16)a3.x; av[13] = (__bf16)a3.y; av[14] = (__bf16)a3.z; av[15] = (__bf16)a3.w;

        const __bf16* bk = bl + (size_t)kt * 512;
        #pragma unroll
        for (int nt = 0; nt < 4; ++nt) {
            v16bf bv = *(const v16bf*)(bk + (size_t)nt * nKt * 512);
            acc[nt] = __builtin_amdgcn_wmma_f32_16x16x32_bf16(
                false, av, false, bv, (short)0, acc[nt], false, false);
        }
    }

    // C/D layout: VGPR r -> row r + 8*(lane/16), col = lane%16 within each tile.
    float* cbase = C + (size_t)(tile * 16 + khalf * 8) * HID + m;
    #pragma unroll
    for (int nt = 0; nt < 4; ++nt) {
        float* crow = cbase + nt * 16;
        #pragma unroll
        for (int r = 0; r < 8; ++r) crow[(size_t)r * HID] = acc[nt][r];
    }
}

// ---------------------------------------------------------------------------
// Attention coefficient projections
// ---------------------------------------------------------------------------
__global__ void k_coef1(const float* __restrict__ Wh,
                        const float* __restrict__ a_s, const float* __restrict__ a_d,
                        float* __restrict__ es, float* __restrict__ ed) {
    int idx = blockIdx.x * blockDim.x + threadIdx.x;
    if (idx >= NNODE * H1) return;
    int n = idx >> 3, h = idx & 7;
    const float* w  = Wh + (size_t)n * HID + h * 8;
    const float* as = a_s + h * 8;
    const float* ad = a_d + h * 8;
    float s = 0.f, d = 0.f;
    #pragma unroll
    for (int o = 0; o < 8; ++o) { float v = w[o]; s += v * as[o]; d += v * ad[o]; }
    es[idx] = s; ed[idx] = d;
}

__global__ void k_coef2(const float* __restrict__ Wh,
                        const float* __restrict__ a_s, const float* __restrict__ a_d,
                        float* __restrict__ es, float* __restrict__ ed) {
    int n = blockIdx.x * blockDim.x + threadIdx.x;
    if (n >= NNODE) return;
    const float* w = Wh + (size_t)n * NLAB;
    float s = 0.f, d = 0.f;
    for (int j = 0; j < NLAB; ++j) { float v = w[j]; s += v * a_s[j]; d += v * a_d[j]; }
    es[n] = s; ed[n] = d;
}

// ---------------------------------------------------------------------------
// Layer-1 edge phase (per edge x head).  Pass 1 materializes the logits
// (streaming write) so passes 2/3 avoid the es/ed random gathers.
// ---------------------------------------------------------------------------
__global__ void k_edge_max1(const int* __restrict__ src, const int* __restrict__ dst,
                            const float* __restrict__ es, const float* __restrict__ ed,
                            float* __restrict__ eatt, unsigned* __restrict__ mk) {
    int idx = blockIdx.x * blockDim.x + threadIdx.x;
    if (idx >= NEDGE * H1) return;
    int e = idx >> 3, h = idx & 7;
    int s = src[e], d = dst[e];
    float v = lrelu(es[s * H1 + h] + ed[d * H1 + h]);
    eatt[idx] = v;
    atomicMax(mk + d * H1 + h, fkey(v));
}

__global__ void k_edge_den1(const int* __restrict__ dst,
                            const float* __restrict__ eatt,
                            const unsigned* __restrict__ mk, float* __restrict__ den) {
    int idx = blockIdx.x * blockDim.x + threadIdx.x;
    if (idx >= NEDGE * H1) return;
    int e = idx >> 3, h = idx & 7;
    int d = dst[e];
    float ex = __expf(eatt[idx] - funkey(mk[d * H1 + h]));
    atomicAdd(den + d * H1 + h, ex);
}

__global__ void k_edge_msg1(const int* __restrict__ src, const int* __restrict__ dst,
                            const float* __restrict__ eatt,
                            const unsigned* __restrict__ mk, const float* __restrict__ den,
                            const float* __restrict__ Wh, float* __restrict__ hacc) {
    int idx = blockIdx.x * blockDim.x + threadIdx.x;
    if (idx >= NEDGE * H1) return;
    int e = idx >> 3, h = idx & 7;
    int s = src[e], d = dst[e];
    float ex = __expf(eatt[idx] - funkey(mk[d * H1 + h]));
    float alpha = ex / (den[d * H1 + h] + 1e-10f);
    const float* w = Wh + (size_t)s * HID + h * 8;
    float4 w0 = *(const float4*)(w);
    float4 w1 = *(const float4*)(w + 4);
    float* o = hacc + (size_t)d * HID + h * 8;
    atomicAdd(o + 0, alpha * w0.x); atomicAdd(o + 1, alpha * w0.y);
    atomicAdd(o + 2, alpha * w0.z); atomicAdd(o + 3, alpha * w0.w);
    atomicAdd(o + 4, alpha * w1.x); atomicAdd(o + 5, alpha * w1.y);
    atomicAdd(o + 6, alpha * w1.z); atomicAdd(o + 7, alpha * w1.w);
}

__global__ void k_elu(float* __restrict__ h) {
    int i = blockIdx.x * blockDim.x + threadIdx.x;
    if (i >= NNODE * HID) return;
    float x = h[i];
    h[i] = x > 0.f ? x : (__expf(x) - 1.f);
}

// ---------------------------------------------------------------------------
// Layer-2 edge phase (H2 == 1)
// ---------------------------------------------------------------------------
__global__ void k_edge_max2(const int* __restrict__ src, const int* __restrict__ dst,
                            const float* __restrict__ es, const float* __restrict__ ed,
                            float* __restrict__ eatt, unsigned* __restrict__ mk) {
    int e = blockIdx.x * blockDim.x + threadIdx.x;
    if (e >= NEDGE) return;
    float v = lrelu(es[src[e]] + ed[dst[e]]);
    eatt[e] = v;
    atomicMax(mk + dst[e], fkey(v));
}

__global__ void k_edge_den2(const int* __restrict__ dst,
                            const float* __restrict__ eatt,
                            const unsigned* __restrict__ mk, float* __restrict__ den) {
    int e = blockIdx.x * blockDim.x + threadIdx.x;
    if (e >= NEDGE) return;
    int d = dst[e];
    atomicAdd(den + d, __expf(eatt[e] - funkey(mk[d])));
}

__global__ void k_edge_msg2(const int* __restrict__ src, const int* __restrict__ dst,
                            const float* __restrict__ eatt,
                            const unsigned* __restrict__ mk, const float* __restrict__ den,
                            const float* __restrict__ Wh, float* __restrict__ out) {
    int idx = blockIdx.x * blockDim.x + threadIdx.x;   // 8 threads per edge
    if (idx >= NEDGE * 8) return;
    int e = idx >> 3, c0 = (idx & 7) * 8;
    int s = src[e], d = dst[e];
    float ex = __expf(eatt[e] - funkey(mk[d]));
    float alpha = ex / (den[d] + 1e-10f);
    const float* w = Wh + (size_t)s * NLAB + c0;
    float4 w0 = *(const float4*)(w);
    float4 w1 = *(const float4*)(w + 4);
    float* o = out + (size_t)d * NLAB + c0;
    atomicAdd(o + 0, alpha * w0.x); atomicAdd(o + 1, alpha * w0.y);
    atomicAdd(o + 2, alpha * w0.z); atomicAdd(o + 3, alpha * w0.w);
    atomicAdd(o + 4, alpha * w1.x); atomicAdd(o + 5, alpha * w1.y);
    atomicAdd(o + 6, alpha * w1.z); atomicAdd(o + 7, alpha * w1.w);
}

// ---------------------------------------------------------------------------
// Row softmax over 64 classes, one wave32 per node, butterfly reductions.
// ---------------------------------------------------------------------------
__global__ void __launch_bounds__(128) k_softmax(float* __restrict__ out) {
    int gid  = blockIdx.x * blockDim.x + threadIdx.x;
    int node = gid >> 5;
    int lane = gid & 31;
    if (node >= NNODE) return;
    float* row = out + (size_t)node * NLAB;
    float x0 = row[lane], x1 = row[lane + 32];
    float mx = fmaxf(x0, x1);
    #pragma unroll
    for (int o = 16; o > 0; o >>= 1) mx = fmaxf(mx, __shfl_xor(mx, o, 32));
    float e0 = __expf(x0 - mx), e1 = __expf(x1 - mx);
    float sum = e0 + e1;
    #pragma unroll
    for (int o = 16; o > 0; o >>= 1) sum += __shfl_xor(sum, o, 32);
    float inv = 1.0f / sum;
    row[lane] = e0 * inv;
    row[lane + 32] = e1 * inv;
}

// ---------------------------------------------------------------------------
// Host-side orchestration
// ---------------------------------------------------------------------------
extern "C" void kernel_launch(void* const* d_in, const int* in_sizes, int n_in,
                              void* d_out, int out_size, void* d_ws, size_t ws_size,
                              hipStream_t stream) {
    const float* X   = (const float*)d_in[0];
    const float* W1  = (const float*)d_in[1];
    const float* a1s = (const float*)d_in[2];
    const float* a1d = (const float*)d_in[3];
    const float* W2  = (const float*)d_in[4];
    const float* a2s = (const float*)d_in[5];
    const float* a2d = (const float*)d_in[6];
    const int*   src = (const int*)d_in[7];
    const int*   dst = (const int*)d_in[8];
    float*       out = (float*)d_out;

    // Workspace carve-out (256B aligned slices)
    char*  ws  = (char*)d_ws;
    size_t off = 0;
    auto alloc = [&](size_t bytes) -> void* {
        void* p = ws + off;
        off = (off + bytes + 255) & ~(size_t)255;
        return p;
    };
    float*    Wh1   = (float*)   alloc((size_t)NNODE * HID  * sizeof(float));
    float*    h1    = (float*)   alloc((size_t)NNODE * HID  * sizeof(float));
    float*    Wh2   = (float*)   alloc((size_t)NNODE * NLAB * sizeof(float));
    float*    es1   = (float*)   alloc((size_t)NNODE * H1   * sizeof(float));
    float*    ed1   = (float*)   alloc((size_t)NNODE * H1   * sizeof(float));
    unsigned* mk1   = (unsigned*)alloc((size_t)NNODE * H1   * sizeof(unsigned));
    float*    den1  = (float*)   alloc((size_t)NNODE * H1   * sizeof(float));
    float*    eatt1 = (float*)   alloc((size_t)NEDGE * H1   * sizeof(float));
    float*    es2   = (float*)   alloc((size_t)NNODE        * sizeof(float));
    float*    ed2   = (float*)   alloc((size_t)NNODE        * sizeof(float));
    unsigned* mk2   = (unsigned*)alloc((size_t)NNODE        * sizeof(unsigned));
    float*    den2  = (float*)   alloc((size_t)NNODE        * sizeof(float));
    float*    eatt2 = (float*)   alloc((size_t)NEDGE        * sizeof(float));
    __bf16*   Bp1   = (__bf16*)  alloc((size_t)4 * 16 * 512 * sizeof(__bf16)); // [nt=4][kt=16][lane=32][e=16]
    __bf16*   Bp2   = (__bf16*)  alloc((size_t)4 * 2  * 512 * sizeof(__bf16)); // [nt=4][kt=2 ][lane=32][e=16]

    const int B = 256;
    auto cdiv = [](long long a, long long b) { return (int)((a + b - 1) / b); };
    const int gemmGrid = cdiv(ROWT, 4);   // 4 waves (row tiles) per 128-thread block

    // Zero accumulators / segment state
    hipMemsetAsync(h1,   0, (size_t)NNODE * HID  * sizeof(float),    stream);
    hipMemsetAsync(mk1,  0, (size_t)NNODE * H1   * sizeof(unsigned), stream);
    hipMemsetAsync(den1, 0, (size_t)NNODE * H1   * sizeof(float),    stream);
    hipMemsetAsync(mk2,  0, (size_t)NNODE        * sizeof(unsigned), stream);
    hipMemsetAsync(den2, 0, (size_t)NNODE        * sizeof(float),    stream);
    hipMemsetAsync(out,  0, (size_t)NNODE * NLAB * sizeof(float),    stream);

    // Weight repack to bf16 WMMA B-operand layout
    k_repack<<<cdiv(4 * 16 * 512, B), B, 0, stream>>>(W1, Bp1, 16, 4 * 16 * 512, 1);
    k_repack<<<cdiv(4 * 2  * 512, B), B, 0, stream>>>(W2, Bp2, 2,  4 * 2  * 512, 0);

    // ---- Layer 1 ----
    k_gemm_bf16<<<gemmGrid, 128, 0, stream>>>(X, FIN, Bp1, FIN / 32, Wh1);
    k_coef1<<<cdiv((long long)NNODE * H1, B), B, 0, stream>>>(Wh1, a1s, a1d, es1, ed1);
    k_edge_max1<<<cdiv((long long)NEDGE * H1, B), B, 0, stream>>>(src, dst, es1, ed1, eatt1, mk1);
    k_edge_den1<<<cdiv((long long)NEDGE * H1, B), B, 0, stream>>>(dst, eatt1, mk1, den1);
    k_edge_msg1<<<cdiv((long long)NEDGE * H1, B), B, 0, stream>>>(src, dst, eatt1, mk1, den1, Wh1, h1);
    k_elu<<<cdiv((long long)NNODE * HID, B), B, 0, stream>>>(h1);

    // ---- Layer 2 ----
    k_gemm_bf16<<<gemmGrid, 128, 0, stream>>>(h1, HID, Bp2, HID / 32, Wh2);
    k_coef2<<<cdiv(NNODE, B), B, 0, stream>>>(Wh2, a2s, a2d, es2, ed2);
    k_edge_max2<<<cdiv(NEDGE, B), B, 0, stream>>>(src, dst, es2, ed2, eatt2, mk2);
    k_edge_den2<<<cdiv(NEDGE, B), B, 0, stream>>>(dst, eatt2, mk2, den2);
    k_edge_msg2<<<cdiv((long long)NEDGE * 8, B), B, 0, stream>>>(src, dst, eatt2, mk2, den2, Wh2, out);

    // ---- Output softmax (mean over H2=1 heads is identity) ----
    k_softmax<<<cdiv((long long)NNODE * 32, 128), 128, 0, stream>>>(out);
}
